// Head_45380624449716
// MI455X (gfx1250) — matrix operand order repaired and checked by reading, not compiled, not run on hardware
//
#include <hip/hip_runtime.h>
#include <hip/hip_bf16.h>

typedef __attribute__((ext_vector_type(16))) __bf16 v16bf;
typedef __attribute__((ext_vector_type(8)))  float  v8f;

#define N_EMBED 1024
#define HEAD    64
#define BATCH   4
#define SEQ     2048
#define ROWS    (BATCH*SEQ)   // 8192 total rows
#define RT      (ROWS/16)     // 512 row tiles
#define QT      (SEQ/16)      // 128 query tiles / batch
#define KB32    (SEQ/32)      // 64 key blocks / batch

#define USE_ASYNC_STAGE 1     // gfx1250 global_load_async_to_lds_b128 staging

__device__ __forceinline__ unsigned short f2bf(float f) {
  unsigned int u = __float_as_uint(f);
  u += 0x7FFFu + ((u >> 16) & 1u);   // round-to-nearest-even
  return (unsigned short)(u >> 16);
}

// ---- pack x (fp32) into WMMA A-fragment-swizzled bf16 -----------------------
// xsw layout: [rowtile 512][cblock 32][lane 32][elem 16]
// A-frag (16x32, bf16): lane l -> M = l&15, kb8 = (l>=16)*8
//   elems 0..7  -> K = kb8..kb8+7 ; elems 8..15 -> K = 16+kb8..16+kb8+7
__global__ void pack_x(const float* __restrict__ x, unsigned short* __restrict__ xsw) {
  int tid  = blockIdx.x * blockDim.x + threadIdx.x;   // < ROWS*N_EMBED
  int e    = tid & 15;
  int lane = (tid >> 4) & 31;
  int cb   = (tid >> 9) & 31;
  int rt   = tid >> 14;
  int row  = rt * 16 + (lane & 15);
  int kb8  = (lane >> 4) * 8;
  int k    = cb * 32 + ((e < 8) ? (kb8 + e) : (16 + kb8 + (e - 8)));
  xsw[tid] = f2bf(x[(size_t)row * N_EMBED + k]);
}

// ---- pack Wq/Wk/Wv into WMMA B-fragment-swizzled bf16 -----------------------
// wsw layout: [i = m*4+nt : 12][cblock 32][lane 32][elem 16]
// B-frag (32x16, bf16): lane l -> N = l&15, kb8 = (l>=16)*8 ; elems 0..7 -> K kb8..kb8+7, 8..15 -> K 16+kb8..16+kb8+7
__global__ void pack_w(const float* __restrict__ Wq, const float* __restrict__ Wk,
                       const float* __restrict__ Wv, unsigned short* __restrict__ wsw) {
  int tid  = blockIdx.x * blockDim.x + threadIdx.x;   // < 3*4*32*512
  int e    = tid & 15;
  int lane = (tid >> 4) & 31;
  int cb   = (tid >> 9) & 31;
  int nt   = (tid >> 14) & 3;
  int m    = tid >> 16;
  const float* W = (m == 0) ? Wq : ((m == 1) ? Wk : Wv);
  int n = nt * 16 + (lane & 15);
  int k = cb * 32 + ((e < 8) ? (((lane >= 16) ? 8 : 0) + e) : (16 + ((lane >= 16) ? 8 : 0) + (e - 8)));
  wsw[tid] = f2bf(W[(size_t)k * HEAD + n]);
}

// ---- QKV projection ---------------------------------------------------------
// 4 waves/workgroup, one 16-row tile each. W fragments for a 4-K-step chunk
// (48 frags = 48 KB) are staged into LDS once per chunk via async DMA and
// shared by all 4 waves; B-frags then come from ds_load_b128 (cheap waits),
// A-frags are pre-issued as one global clause per chunk.
__global__ void __launch_bounds__(128) proj_qkv(
    const unsigned short* __restrict__ xsw,
    const unsigned short* __restrict__ wsw,
    unsigned short* __restrict__ Qsw,
    unsigned short* __restrict__ Ksw,
    unsigned short* __restrict__ Vsw) {
  __shared__ unsigned short smem[48 * 512];   // [cbl 4][i 12][512] = 48 KB
  int lane = threadIdx.x & 31;
  int rt   = blockIdx.x * 4 + (threadIdx.x >> 5);   // 0..511

  v8f acc[3][4];
  #pragma unroll
  for (int m = 0; m < 3; ++m)
    #pragma unroll
    for (int nt = 0; nt < 4; ++nt) acc[m][nt] = (v8f){};

  for (int chunk = 0; chunk < 8; ++chunk) {
    int cb0 = chunk * 4;

    // ---- stage W fragments for this chunk into LDS --------------------------
    // 48 frags x 64 x 16B units = 3072 units; 24 per thread.
#if USE_ASYNC_STAGE
    {
      unsigned int lds_base = (unsigned int)(size_t)&smem[0];
      #pragma unroll
      for (int j = 0; j < 24; ++j) {
        int u   = threadIdx.x + 128 * j;
        int f   = u >> 6;            // fragment 0..47  (f = cbl*12 + i)
        int w   = u & 63;            // 16B unit within fragment
        int cbl = f / 12, i = f - cbl * 12;
        unsigned long long gaddr =
            (unsigned long long)(size_t)(wsw + ((size_t)(i * 32 + cb0 + cbl) * 512 + w * 8));
        unsigned int laddr = lds_base + (unsigned int)((f * 512 + w * 8) * 2);
        asm volatile("global_load_async_to_lds_b128 %0, %1, off"
                     :: "v"(laddr), "v"(gaddr) : "memory");
      }
      asm volatile("s_wait_asynccnt 0x0" ::: "memory");
    }
#else
    {
      #pragma unroll
      for (int j = 0; j < 24; ++j) {
        int u   = threadIdx.x + 128 * j;
        int f   = u >> 6;
        int w   = u & 63;
        int cbl = f / 12, i = f - cbl * 12;
        const uint4* src = (const uint4*)(wsw + ((size_t)(i * 32 + cb0 + cbl) * 512 + w * 8));
        *(uint4*)(smem + (size_t)(f * 512 + w * 8)) = *src;
      }
    }
#endif
    __syncthreads();

    // ---- pre-issue the 4 A fragments for this chunk (one load clause) -------
    v16bf a4[4];
    #pragma unroll
    for (int cbl = 0; cbl < 4; ++cbl)
      a4[cbl] = *(const v16bf*)(xsw + ((size_t)(rt * 32 + cb0 + cbl) * 32 + lane) * 16);
    if (chunk + 1 < 8)
      __builtin_prefetch(xsw + ((size_t)(rt * 32 + cb0 + 4) * 32 + lane) * 16, 0, 1);

    // ---- 48 WMMAs against LDS-resident B fragments --------------------------
    #pragma unroll
    for (int cbl = 0; cbl < 4; ++cbl) {
      #pragma unroll
      for (int i = 0; i < 12; ++i) {
        v16bf bw = *(const v16bf*)(smem + (size_t)((cbl * 12 + i) * 512 + lane * 16));
        acc[i >> 2][i & 3] = __builtin_amdgcn_wmma_f32_16x16x32_bf16(
            false, a4[cbl], false, bw, (short)0, acc[i >> 2][i & 3], false, false);
      }
    }
    __syncthreads();
  }

  // C-frag: lane l holds N = l&15 (col/h), rows M = r + (l>=16)*8
  int col   = lane & 15;
  int rhalf = (lane >> 4) * 8;
  #pragma unroll
  for (int nt = 0; nt < 4; ++nt) {
    #pragma unroll
    for (int r = 0; r < 8; ++r) {
      int grow = rt * 16 + rhalf + r;
      int b = grow >> 11;          // / SEQ
      int t = grow & 2047;
      int h = nt * 16 + col;
      { // Q -> B-frag swizzle (lane = query%16 + 16*lhalf, split-8 elems over h)
        int qtile = t >> 4, qi = t & 15;
        int hb = h >> 5, hr = h & 31;
        int lp = qi + (((hr >> 3) & 1) << 4);
        int el = (hr & 7) + ((hr >= 16) ? 8 : 0);
        Qsw[((size_t)((b * QT + qtile) * 2 + hb) * 32 + lp) * 16 + el] = f2bf(acc[0][nt][r]);
      }
      { // K -> A-frag swizzle (lane = key%16 + 16*lhalf, split-8 elems over h)
        int kt = t >> 4, ki = t & 15;
        int hb = h >> 5, hr = h & 31;
        int lhalf = (hr >> 3) & 1;
        int el = (hr & 7) + ((hr >= 16) ? 8 : 0);
        int lp = ki + (lhalf << 4);
        Ksw[((size_t)((b * QT + kt) * 2 + hb) * 32 + lp) * 16 + el] = f2bf(acc[1][nt][r]);
      }
      { // V -> V^T A-frag swizzle (M = h, K = key)
        int kb = t >> 5, kr = t & 31;
        int ht = h >> 4, hi = h & 15;
        int lhalf = kr >> 4;
        int el = kr & 15;
        int lp = hi + (lhalf << 4);
        Vsw[((size_t)((b * KB32 + kb) * 4 + ht) * 32 + lp) * 16 + el] = f2bf(acc[2][nt][r]);
      }
    }
  }
}

// ---- flash attention, transposed (S^T = K Q^T) so softmax is per-lane -------
__global__ void __launch_bounds__(128) attn(
    const unsigned short* __restrict__ Qsw,
    const unsigned short* __restrict__ Ksw,
    const unsigned short* __restrict__ Vsw,
    float* __restrict__ out) {
  int lane = threadIdx.x & 31;
  int gw   = blockIdx.x * 4 + (threadIdx.x >> 5);  // 0..511
  int b    = gw >> 7;
  int qt   = gw & 127;

  v16bf qf0 = *(const v16bf*)(Qsw + ((size_t)((b * QT + qt) * 2 + 0) * 32 + lane) * 16);
  v16bf qf1 = *(const v16bf*)(Qsw + ((size_t)((b * QT + qt) * 2 + 1) * 32 + lane) * 16);

  v8f acc[4];
  #pragma unroll
  for (int ht = 0; ht < 4; ++ht) acc[ht] = (v8f){};

  float mrun = -3.0e38f, ssum = 0.0f;
  int   query   = qt * 16 + (lane & 15);
  int   krowoff = (lane >> 4) * 8;
  const float scale = 0.03125f;   // C^-0.5, C = 1024
  int kbmax = (qt * 16 + 15) >> 5;

  const unsigned short* Kbase = Ksw + (size_t)(b * QT) * 2 * 512;
  const unsigned short* Vbase = Vsw + (size_t)(b * KB32) * 4 * 512;

  // preload K fragments for kb = 0 (key tiles 0 and 1, h-blocks 0 and 1)
  v16bf ka[4];
  #pragma unroll
  for (int i = 0; i < 4; ++i)   // i = kt*2 + hb
    ka[i] = *(const v16bf*)(Kbase + ((size_t)i * 32 + lane) * 16);

  for (int kb = 0; kb <= kbmax; ++kb) {
    // V fragments for this block: issue early, consumed after softmax
    v16bf va[4];
    #pragma unroll
    for (int ht = 0; ht < 4; ++ht)
      va[ht] = *(const v16bf*)(Vbase + ((size_t)(kb * 4 + ht) * 32 + lane) * 16);

    // S^T tiles from the pre-loaded K fragments
    v8f st0 = (v8f){}, st1 = (v8f){};
    st0 = __builtin_amdgcn_wmma_f32_16x16x32_bf16(false, ka[0], false, qf0, (short)0, st0, false, false);
    st0 = __builtin_amdgcn_wmma_f32_16x16x32_bf16(false, ka[1], false, qf1, (short)0, st0, false, false);
    st1 = __builtin_amdgcn_wmma_f32_16x16x32_bf16(false, ka[2], false, qf0, (short)0, st1, false, false);
    st1 = __builtin_amdgcn_wmma_f32_16x16x32_bf16(false, ka[3], false, qf1, (short)0, st1, false, false);

    // prefetch next iteration's K fragments; latency hidden by softmax below
    if (kb < kbmax) {
      const unsigned short* kn = Kbase + (size_t)(kb + 1) * 4 * 512;
      __builtin_prefetch(kn + (size_t)4 * 512, 0, 1);
      #pragma unroll
      for (int i = 0; i < 4; ++i)
        ka[i] = *(const v16bf*)(kn + ((size_t)i * 32 + lane) * 16);
    }

    // scale + causal mask + block max (per lane; merge lane pair via xor-16)
    float p0[8], p1[8], bm = -3.0e38f;
    #pragma unroll
    for (int r = 0; r < 8; ++r) {
      int key0 = (kb * 2) * 16 + krowoff + r;
      int key1 = (kb * 2 + 1) * 16 + krowoff + r;
      float s0 = (key0 <= query) ? st0[r] * scale : -3.0e38f;
      float s1 = (key1 <= query) ? st1[r] * scale : -3.0e38f;
      p0[r] = s0; p1[r] = s1;
      bm = fmaxf(bm, fmaxf(s0, s1));
    }
    bm = fmaxf(bm, __shfl_xor(bm, 16, 32));
    float mnew  = fmaxf(mrun, bm);
    float alpha = __expf(mrun - mnew);
    float partial = 0.0f;
    #pragma unroll
    for (int r = 0; r < 8; ++r) {
      p0[r] = __expf(p0[r] - mnew);
      p1[r] = __expf(p1[r] - mnew);
      partial += p0[r] + p1[r];
    }
    ssum = ssum * alpha + partial + __shfl_xor(partial, 16, 32);
    mrun = mnew;
    #pragma unroll
    for (int ht = 0; ht < 4; ++ht)
      #pragma unroll
      for (int r = 0; r < 8; ++r) acc[ht][r] *= alpha;

    // Build P^T B-frag (lane = query, elems = 16 contiguous keys of the 32-block)
    union { v16bf v; unsigned short u[16]; } bp;
    bool hi = lane >= 16;
    #pragma unroll
    for (int r = 0; r < 8; ++r) {
      float o0 = __shfl_xor(p0[r], 16, 32);   // keys 8..15 (tile0) from partner
      float o1 = __shfl_xor(p1[r], 16, 32);   // keys 16..23 (tile1) from partner
      bp.u[r]     = f2bf(hi ? o1    : p0[r]);
      bp.u[8 + r] = f2bf(hi ? p1[r] : o0);
    }

    // O^T += V^T (A-frag) x P^T (B-frag)
    #pragma unroll
    for (int ht = 0; ht < 4; ++ht)
      acc[ht] = __builtin_amdgcn_wmma_f32_16x16x32_bf16(
          false, va[ht], false, bp.v, (short)0, acc[ht], false, false);
  }

  float rs = 1.0f / ssum;
  #pragma unroll
  for (int ht = 0; ht < 4; ++ht) {
    #pragma unroll
    for (int r = 0; r < 8; ++r) {
      int h = ht * 16 + krowoff + r;
      out[((size_t)(b * SEQ + query)) * HEAD + h] = acc[ht][r] * rs;
    }
  }
}

extern "C" void kernel_launch(void* const* d_in, const int* in_sizes, int n_in,
                              void* d_out, int out_size, void* d_ws, size_t ws_size,
                              hipStream_t stream) {
  const float* x  = (const float*)d_in[0];
  const float* Wq = (const float*)d_in[1];
  const float* Wk = (const float*)d_in[2];
  const float* Wv = (const float*)d_in[3];
  float* out = (float*)d_out;

  unsigned short* ws  = (unsigned short*)d_ws;
  unsigned short* xsw = ws;                                       // 8,388,608
  unsigned short* wsw = xsw + (size_t)ROWS * N_EMBED;             //   196,608
  unsigned short* Qsw = wsw + (size_t)3 * 4 * 32 * 512;           //   524,288
  unsigned short* Ksw = Qsw + (size_t)BATCH * QT * 2 * 512;       //   524,288
  unsigned short* Vsw = Ksw + (size_t)BATCH * QT * 2 * 512;       //   524,288

  pack_x<<<(ROWS * N_EMBED) / 256, 256, 0, stream>>>(x, xsw);
  pack_w<<<(3 * 4 * 32 * 512) / 256, 256, 0, stream>>>(Wq, Wk, Wv, wsw);
  proj_qkv<<<RT / 4, 128, 0, stream>>>(xsw, wsw, Qsw, Ksw, Vsw);
  attn<<<(BATCH * QT) / 4, 128, 0, stream>>>(Qsw, Ksw, Vsw, out);
}
